// DeformableCrossAttention2D_K_84189948936663
// MI455X (gfx1250) — matrix operand order repaired and checked by reading, not compiled, not run on hardware
//
#include <hip/hip_runtime.h>

#define DIMX   256
#define DH     64
#define GG     8
#define IN_G   32
#define OUT_G  64
#define POS_H  64
#define INNER  512
#define HH     256
#define WW     256
#define HOW    64
#define NPIX   4096
#define BATCH  4
#define KQ     64
#define BGC    32
#define KTOT   1152   // 36 taps * 32 ch
#define NPART  8
#define QSCALE 0.125f // 64^-0.5
#define KSTR   72     // Ks LDS row stride (halves), 144B = 16B aligned
#define VSTR   136    // Vs LDS row stride (halves), 272B = 16B aligned

typedef _Float16 h16;
typedef __attribute__((ext_vector_type(16))) _Float16 v16h;
typedef __attribute__((ext_vector_type(8)))  float    v8f;

__device__ __forceinline__ v8f wmma_f16f32(v16h a, v16h b, v8f c) {
  return __builtin_amdgcn_wmma_f32_16x16x32_f16(false, a, false, b, (short)0, c, false, false);
}

// Async global -> LDS 128-bit copy (CDNA5 TDM-adjacent async path, ASYNCcnt-tracked).
__device__ __forceinline__ void async_ld_b128(uint32_t lds_off, const void* gptr) {
  asm volatile("global_load_async_to_lds_b128 %0, %1, off"
               :: "v"(lds_off), "v"(gptr) : "memory");
}
__device__ __forceinline__ void wait_async0() {
  asm volatile("s_wait_asynccnt 0x0" ::: "memory");
}

// A fragment (16x32 f16), source row-major [m][k] with k contiguous, row stride ldk (halves).
// ISA layout: lanes 0-15 m=0..15; vgpr j<4: k = hi*8 + 2j; vgpr j>=4: k = 16 + hi*8 + 2(j-4).
__device__ __forceinline__ v16h frag_a_f16(const h16* base, int ldk) {
  const int l = threadIdx.x & 31;
  const h16* pm = base + (l & 15) * ldk;
  const int kb = (l >> 4) * 8;
  v16h a;
#pragma unroll
  for (int j = 0; j < 4; ++j) {
    a[2*j]         = pm[kb + 2*j];
    a[2*j + 1]     = pm[kb + 2*j + 1];
    a[8 + 2*j]     = pm[16 + kb + 2*j];
    a[8 + 2*j + 1] = pm[16 + kb + 2*j + 1];
  }
  return a;
}

// B fragment (32x16 f16), source [n][k] layout (k contiguous), stride ldk.
// ISA layout: lane = n; vgpr j<4: k = hi*8 + 2j; vgpr j>=4: k = 16 + hi*8 + 2(j-4).
__device__ __forceinline__ v16h frag_b_f16(const h16* base, int ldk) {
  const int l = threadIdx.x & 31;
  const h16* pn = base + (l & 15) * ldk;
  const int kb = (l >> 4) * 8;
  v16h b;
#pragma unroll
  for (int j = 0; j < 4; ++j) {
    b[2*j]         = pn[kb + 2*j];
    b[2*j + 1]     = pn[kb + 2*j + 1];
    b[8 + 2*j]     = pn[16 + kb + 2*j];
    b[8 + 2*j + 1] = pn[16 + kb + 2*j + 1];
  }
  return b;
}

// ---------------- K0: weight prep (combined offsets weight + f16 layouts) ----------------
__global__ void k_prep(const float* __restrict__ Wtk, const float* __restrict__ Wdw,
                       const float* __restrict__ Wk,  const float* __restrict__ Wv,
                       const float* __restrict__ Wp2, const float* __restrict__ Wp3,
                       const float* __restrict__ Wp4,
                       h16* __restrict__ W2h, h16* __restrict__ Wkh, h16* __restrict__ Wvh,
                       h16* __restrict__ Wp2T, h16* __restrict__ Wp3T, h16* __restrict__ Wp4T) {
  int stride = gridDim.x * blockDim.x;
  int tid = blockIdx.x * blockDim.x + threadIdx.x;
  for (int idx = tid; idx < GG*OUT_G*KTOT; idx += stride) {
    int k = idx % KTOT; int c = (idx / KTOT) % OUT_G; int g = idx / (KTOT*OUT_G);
    int tap = k >> 5, i = k & 31;
    W2h[idx] = (h16)(Wtk[(g*OUT_G + c)*IN_G + i] * Wdw[c*36 + tap]);
  }
  for (int idx = tid; idx < GG*OUT_G*IN_G; idx += stride) {
    Wkh[idx] = (h16)Wk[idx];   // already [g][o][i] == [n][k]
    Wvh[idx] = (h16)Wv[idx];
  }
  for (int idx = tid; idx < POS_H*POS_H; idx += stride) {
    int n = idx >> 6, k = idx & 63;
    Wp2T[idx] = (h16)Wp2[k*POS_H + n];
    Wp3T[idx] = (h16)Wp3[k*POS_H + n];
  }
  for (int idx = tid; idx < IN_G*POS_H; idx += stride) {
    int n = idx >> 6, k = idx & 63;       // n<32, k<64
    Wp4T[idx] = (h16)Wp4[k*IN_G + n];
  }
}

// ---------------- K1: fused grouped-1x1 + depthwise 6x6/s4 + GELU + pw + tanh -> vgrid ----
#define PSTR 33
__global__ __launch_bounds__(128) void k_offsets(
    const float* __restrict__ bev, const h16* __restrict__ W2h,
    const float* __restrict__ bdw, const float* __restrict__ Wpw,
    float* __restrict__ vgrid) {
  __shared__ h16 patch[18*18*PSTR];
  __shared__ float offc[16][65];
  int bg = blockIdx.y; int b = bg >> 3, g = bg & 7;
  int tY = blockIdx.x >> 4, tX = blockIdx.x & 15;
  int y0 = tY*16 - 1, x0 = tX*16 - 1;   // pad = 1
  const float* bevg = bev + ((size_t)(b*DIMX + g*IN_G)) * (HH*WW);
  for (int i = 0; i < IN_G; ++i) {
    const float* ch = bevg + (size_t)i*(HH*WW);
    for (int p = threadIdx.x; p < 18*18; p += 128) {
      int ly = p / 18, lx = p - ly*18;
      int y = y0 + ly, x = x0 + lx;
      float v = 0.f;
      if ((unsigned)y < 256u && (unsigned)x < 256u) v = ch[y*WW + x];
      patch[p*PSTR + i] = (h16)v;
    }
  }
  __syncthreads();
  int wave = threadIdx.x >> 5;
  int l = threadIdx.x & 31;
  const h16* Bbase = W2h + (size_t)g*OUT_G*KTOT + (size_t)(wave*16)*KTOT;
  v8f acc = {0.f,0.f,0.f,0.f,0.f,0.f,0.f,0.f};
  int m = l & 15;
  int ty = m >> 2, tx = m & 3;
  int kb = (l >> 4) * 8;
  for (int s = 0; s < 36; ++s) {
    int ky = s / 6, kx = s - ky*6;
    const h16* pm = &patch[((ty*4 + ky)*18 + (tx*4 + kx))*PSTR];
    v16h a;
#pragma unroll
    for (int j = 0; j < 4; ++j) {
      a[2*j]       = pm[kb + 2*j];
      a[2*j+1]     = pm[kb + 2*j + 1];
      a[8 + 2*j]   = pm[16 + kb + 2*j];
      a[8 + 2*j+1] = pm[16 + kb + 2*j + 1];
    }
    v16h bf = frag_b_f16(Bbase + s*32, KTOT);
    acc = wmma_f16f32(a, bf, acc);
  }
#pragma unroll
  for (int r = 0; r < 8; ++r)
    offc[r + ((l >> 4) << 3)][wave*16 + (l & 15)] = acc[r];
  __syncthreads();
  if (threadIdx.x < 16) {
    int mm = threadIdx.x;
    float o0 = 0.f, o1 = 0.f;
    for (int c = 0; c < OUT_G; ++c) {
      float xg = offc[mm][c] + bdw[c];
      float ge = 0.5f * xg * (1.f + erff(xg * 0.70710678118f));  // exact GELU
      o0 += Wpw[c] * ge;
      o1 += Wpw[OUT_G + c] * ge;
    }
    float offx = 4.f * tanhf(o0);
    float offy = 4.f * tanhf(o1);
    int ho = tY*4 + (mm >> 2), wo = tX*4 + (mm & 3);
    float vxs = 2.f * ((float)wo + offx) / 63.f - 1.f;
    float vys = 2.f * ((float)ho + offy) / 63.f - 1.f;
    size_t base = (((size_t)bg*NPIX) + ho*HOW + wo) * 2;
    vgrid[base] = vxs; vgrid[base + 1] = vys;
  }
}

// ---------------- K2: bilinear sample + pos-MLP (WMMA) + k/v projections (WMMA) ----------
__global__ __launch_bounds__(128) void k_sample_kv(
    const float* __restrict__ bev, const float* __restrict__ vgrid,
    const float* __restrict__ Wp1, const float* __restrict__ bp1,
    const float* __restrict__ bp2, const float* __restrict__ bp3,
    const float* __restrict__ bp4,
    const h16* __restrict__ Wp2T, const h16* __restrict__ Wp3T,
    const h16* __restrict__ Wp4T,
    const h16* __restrict__ Wkh, const h16* __restrict__ Wvh,
    h16* __restrict__ kh, h16* __restrict__ vh) {
  __shared__ float samp[64*33];
  __shared__ h16 hA[64*66];
  __shared__ h16 hB[64*66];
  __shared__ h16 kvs[64*34];
  __shared__ float vgl[64][2];
  int bg = blockIdx.y, ho = blockIdx.x;
  int b = bg >> 3, g = bg & 7;
  if (threadIdx.x < 64) {
    size_t base = (((size_t)bg*NPIX) + ho*HOW + threadIdx.x)*2;
    vgl[threadIdx.x][0] = vgrid[base];
    vgl[threadIdx.x][1] = vgrid[base + 1];
  }
  __syncthreads();
  {
    int pix = threadIdx.x >> 1;
    int ioff = (threadIdx.x & 1) * 16;
    float xs = vgl[pix][0], ys = vgl[pix][1];
    float x = (xs + 1.f) * 128.f - 0.5f;
    float y = (ys + 1.f) * 128.f - 0.5f;
    float xf = floorf(x), yf = floorf(y);
    int ix0 = (int)xf, iy0 = (int)yf;
    float fx = x - xf, fy = y - yf;
    float w00 = (1.f-fx)*(1.f-fy), w10 = fx*(1.f-fy), w01 = (1.f-fx)*fy, w11 = fx*fy;
    if (!((unsigned)ix0     < 256u)) { w00 = 0.f; w01 = 0.f; }
    if (!((unsigned)(ix0+1) < 256u)) { w10 = 0.f; w11 = 0.f; }
    if (!((unsigned)iy0     < 256u)) { w00 = 0.f; w10 = 0.f; }
    if (!((unsigned)(iy0+1) < 256u)) { w01 = 0.f; w11 = 0.f; }
    int x0c = ix0 < 0 ? 0 : (ix0 > 255 ? 255 : ix0);
    int x1c = ix0+1 < 0 ? 0 : (ix0+1 > 255 ? 255 : ix0+1);
    int y0c = iy0 < 0 ? 0 : (iy0 > 255 ? 255 : iy0);
    int y1c = iy0+1 < 0 ? 0 : (iy0+1 > 255 ? 255 : iy0+1);
    const float* bevg = bev + ((size_t)(b*DIMX + g*IN_G + ioff)) * (HH*WW);
#pragma unroll 4
    for (int i = 0; i < 16; ++i) {
      const float* ch = bevg + (size_t)i*(HH*WW);
      float v = w00*ch[y0c*WW + x0c] + w10*ch[y0c*WW + x1c]
              + w01*ch[y1c*WW + x0c] + w11*ch[y1c*WW + x1c];
      samp[pix*33 + ioff + i] = v;
    }
  }
  { // pos-MLP layer 1 (K=2, scalar)
    int pix = threadIdx.x >> 1;
    int j0 = (threadIdx.x & 1) * 32;
    float xs = vgl[pix][0], ys = vgl[pix][1];
#pragma unroll
    for (int j = 0; j < 32; ++j) {
      float hv = xs*Wp1[j0 + j] + ys*Wp1[POS_H + j0 + j] + bp1[j0 + j];
      hA[pix*66 + j0 + j] = (h16)fmaxf(hv, 0.f);
    }
  }
  __syncthreads();
  int wave = threadIdx.x >> 5, l = threadIdx.x & 31;
  const v8f vzero = {0.f,0.f,0.f,0.f,0.f,0.f,0.f,0.f};
  { // layer 2
    v8f acc[4] = {vzero, vzero, vzero, vzero};
#pragma unroll
    for (int s = 0; s < 2; ++s) {
      v16h a = frag_a_f16(hA + (wave*16)*66 + s*32, 66);
#pragma unroll
      for (int nt = 0; nt < 4; ++nt)
        acc[nt] = wmma_f16f32(a, frag_b_f16(Wp2T + (nt*16)*POS_H + s*32, POS_H), acc[nt]);
    }
#pragma unroll
    for (int nt = 0; nt < 4; ++nt)
#pragma unroll
      for (int r = 0; r < 8; ++r) {
        int mm = wave*16 + r + ((l >> 4) << 3);
        int n = nt*16 + (l & 15);
        hB[mm*66 + n] = (h16)fmaxf(acc[nt][r] + bp2[n], 0.f);
      }
  }
  __syncthreads();
  { // layer 3
    v8f acc[4] = {vzero, vzero, vzero, vzero};
#pragma unroll
    for (int s = 0; s < 2; ++s) {
      v16h a = frag_a_f16(hB + (wave*16)*66 + s*32, 66);
#pragma unroll
      for (int nt = 0; nt < 4; ++nt)
        acc[nt] = wmma_f16f32(a, frag_b_f16(Wp3T + (nt*16)*POS_H + s*32, POS_H), acc[nt]);
    }
#pragma unroll
    for (int nt = 0; nt < 4; ++nt)
#pragma unroll
      for (int r = 0; r < 8; ++r) {
        int mm = wave*16 + r + ((l >> 4) << 3);
        int n = nt*16 + (l & 15);
        hA[mm*66 + n] = (h16)fmaxf(acc[nt][r] + bp3[n], 0.f);
      }
  }
  __syncthreads();
  { // layer 4 + add sampled -> kv (f16)
    v8f acc[2] = {vzero, vzero};
#pragma unroll
    for (int s = 0; s < 2; ++s) {
      v16h a = frag_a_f16(hA + (wave*16)*66 + s*32, 66);
#pragma unroll
      for (int nt = 0; nt < 2; ++nt)
        acc[nt] = wmma_f16f32(a, frag_b_f16(Wp4T + (nt*16)*POS_H + s*32, POS_H), acc[nt]);
    }
#pragma unroll
    for (int nt = 0; nt < 2; ++nt)
#pragma unroll
      for (int r = 0; r < 8; ++r) {
        int mm = wave*16 + r + ((l >> 4) << 3);
        int n = nt*16 + (l & 15);
        kvs[mm*34 + n] = (h16)(acc[nt][r] + bp4[n] + samp[mm*33 + n]);
      }
  }
  __syncthreads();
  { // k and v grouped projections (K=32 == one WMMA step)
    v16h a = frag_a_f16(kvs + (wave*16)*34, 34);
    const h16* Wkg = Wkh + (size_t)g*OUT_G*IN_G;
    const h16* Wvg = Wvh + (size_t)g*OUT_G*IN_G;
    h16* kout = kh + (((size_t)bg*NPIX) + ho*HOW) * DH;   // [bg][n][dh]
    h16* vout = vh + (size_t)bg*DH*NPIX;                  // [bg][dh][n]
    v8f acck[4] = {vzero,vzero,vzero,vzero};
    v8f accv[4] = {vzero,vzero,vzero,vzero};
#pragma unroll
    for (int nt = 0; nt < 4; ++nt) {
      acck[nt] = wmma_f16f32(a, frag_b_f16(Wkg + (nt*16)*IN_G, IN_G), acck[nt]);
      accv[nt] = wmma_f16f32(a, frag_b_f16(Wvg + (nt*16)*IN_G, IN_G), accv[nt]);
    }
#pragma unroll
    for (int nt = 0; nt < 4; ++nt)
#pragma unroll
      for (int r = 0; r < 8; ++r) {
        int mm = wave*16 + r + ((l >> 4) << 3);
        int n = nt*16 + (l & 15);
        kout[(size_t)mm*DH + n] = (h16)acck[nt][r];
        vout[(size_t)n*NPIX + ho*HOW + mm] = (h16)accv[nt][r];
      }
  }
}

// ---------------- K3: q projection (scalar, 67 MFLOP) ----------------
__global__ void k_qproj(const float* __restrict__ mq, const float* __restrict__ Wq,
                        const float* __restrict__ bq, h16* __restrict__ qh) {
  int idx = blockIdx.x * blockDim.x + threadIdx.x;
  if (idx >= BATCH*KQ*INNER) return;
  int c = idx % INNER; int q = (idx / INNER) % KQ; int b = idx / (INNER*KQ);
  const float* qrow = mq + ((size_t)q*BATCH + b)*DIMX;   // q_in[b][q][:] = mq[q][b][:]
  float acc = bq[c];
  for (int d = 0; d < DIMX; ++d) acc += qrow[d] * Wq[d*INNER + c];
  acc *= QSCALE;
  int g = c >> 6, dh = c & 63;
  qh[(((size_t)b*GG + g)*KQ + q)*DH + dh] = (h16)acc;
}

// ---------------- K4: flash attention, key-partitioned, async K/V staging ----------------
__global__ __launch_bounds__(128) void k_attn(
    const h16* __restrict__ qh, const h16* __restrict__ kh,
    const h16* __restrict__ vh, float* __restrict__ opart,
    float* __restrict__ ml) {
  __shared__ __align__(16) h16 Ks[128*KSTR];
  __shared__ __align__(16) h16 Vs[64*VSTR];
  __shared__ h16 Pb[4*16*130];
  int bg = blockIdx.x, p = blockIdx.y;
  int wave = threadIdx.x >> 5, l = threadIdx.x & 31;
  const h16* qbase = qh + ((size_t)bg*KQ + wave*16) * DH;
  v16h aq0 = frag_a_f16(qbase, DH);
  v16h aq1 = frag_a_f16(qbase + 32, DH);
  const v8f vzero = {0.f,0.f,0.f,0.f,0.f,0.f,0.f,0.f};
  v8f o[4] = {vzero,vzero,vzero,vzero};
  float mrow[8], lrow[8];
#pragma unroll
  for (int r = 0; r < 8; ++r) { mrow[r] = -1e30f; lrow[r] = 0.f; }
  const h16* kg = kh + (size_t)bg*NPIX*DH;
  const h16* vg = vh + (size_t)bg*DH*NPIX;
  uint32_t KsBase = (uint32_t)(uintptr_t)&Ks[0];
  uint32_t VsBase = (uint32_t)(uintptr_t)&Vs[0];
  for (int cc = 0; cc < 4; ++cc) {
    int c0 = p*512 + cc*128;
    __syncthreads();
    // Ks: 128 rows x 128B, async b128 copies (8 segs/row, 8 per thread)
#pragma unroll
    for (int t = 0; t < 8; ++t) {
      int idx = threadIdx.x + t*128;
      int row = idx >> 3, seg = idx & 7;
      async_ld_b128(KsBase + (uint32_t)(row*(KSTR*2) + seg*16),
                    (const char*)(kg + (size_t)(c0 + row)*DH) + seg*16);
    }
    // Vs: 64 rows x 256B, async b128 copies (16 segs/row, 8 per thread)
#pragma unroll
    for (int t = 0; t < 8; ++t) {
      int idx = threadIdx.x + t*128;
      int row = idx >> 4, seg = idx & 15;
      async_ld_b128(VsBase + (uint32_t)(row*(VSTR*2) + seg*16),
                    (const char*)(vg + (size_t)row*NPIX + c0) + seg*16);
    }
    wait_async0();
    __syncthreads();
    v8f s[8];
#pragma unroll
    for (int nt = 0; nt < 8; ++nt) {
      s[nt] = vzero;
      s[nt] = wmma_f16f32(aq0, frag_b_f16(Ks + (nt*16)*KSTR, KSTR), s[nt]);
      s[nt] = wmma_f16f32(aq1, frag_b_f16(Ks + (nt*16)*KSTR + 32, KSTR), s[nt]);
    }
#pragma unroll
    for (int r = 0; r < 8; ++r) {
      float mx = s[0][r];
#pragma unroll
      for (int nt = 1; nt < 8; ++nt) mx = fmaxf(mx, s[nt][r]);
      mx = fmaxf(mx, __shfl_xor(mx, 1));
      mx = fmaxf(mx, __shfl_xor(mx, 2));
      mx = fmaxf(mx, __shfl_xor(mx, 4));
      mx = fmaxf(mx, __shfl_xor(mx, 8));   // half-wave: rows differ between lane halves
      float mn = fmaxf(mrow[r], mx);
      float corr = __expf(mrow[r] - mn);
      int row = r + ((l >> 4) << 3);
      float psum = 0.f;
#pragma unroll
      for (int nt = 0; nt < 8; ++nt) {
        float pv = __expf(s[nt][r] - mn);
        psum += pv;
        Pb[wave*2080 + row*130 + nt*16 + (l & 15)] = (h16)pv;
      }
      psum += __shfl_xor(psum, 1);
      psum += __shfl_xor(psum, 2);
      psum += __shfl_xor(psum, 4);
      psum += __shfl_xor(psum, 8);
      lrow[r] = lrow[r]*corr + psum;
      mrow[r] = mn;
#pragma unroll
      for (int nt = 0; nt < 4; ++nt) o[nt][r] *= corr;
    }
    __syncthreads();   // publish Pb (and keep waves lockstep for Ks/Vs reuse)
#pragma unroll
    for (int s4 = 0; s4 < 4; ++s4) {
      v16h ap = frag_a_f16(Pb + wave*2080 + s4*32, 130);
#pragma unroll
      for (int nt = 0; nt < 4; ++nt)
        o[nt] = wmma_f16f32(ap, frag_b_f16(Vs + (nt*16)*VSTR + s4*32, VSTR), o[nt]);
    }
  }
  float* ob = opart + (((size_t)bg*NPART + p)*KQ) * DH;
#pragma unroll
  for (int nt = 0; nt < 4; ++nt)
#pragma unroll
    for (int r = 0; r < 8; ++r) {
      int mm = wave*16 + r + ((l >> 4) << 3);
      int n = nt*16 + (l & 15);
      ob[(size_t)mm*DH + n] = o[nt][r];
    }
  if ((l & 15) == 0) {
#pragma unroll
    for (int r = 0; r < 8; ++r) {
      int mm = wave*16 + r + ((l >> 4) << 3);
      size_t base = (((size_t)bg*NPART + p)*KQ + mm)*2;
      ml[base] = mrow[r];
      ml[base + 1] = lrow[r];
    }
  }
}

// ---------------- K5: LSE-merge of key partitions ----------------
__global__ void k_attn_reduce(const float* __restrict__ opart, const float* __restrict__ ml,
                              float* __restrict__ oattn) {
  int idx = blockIdx.x * blockDim.x + threadIdx.x;
  if (idx >= BGC*KQ*DH) return;
  int dh = idx & 63; int q = (idx >> 6) & 63; int bg = idx >> 12;
  float M = -1e30f;
  for (int p = 0; p < NPART; ++p)
    M = fmaxf(M, ml[(((size_t)bg*NPART + p)*KQ + q)*2]);
  float L = 0.f, acc = 0.f;
  for (int p = 0; p < NPART; ++p) {
    size_t base = (((size_t)bg*NPART + p)*KQ + q)*2;
    float w = __expf(ml[base] - M);
    L += ml[base + 1] * w;
    acc += opart[(((size_t)bg*NPART + p)*KQ + q)*DH + dh] * w;
  }
  oattn[((size_t)bg*KQ + q)*DH + dh] = acc / L;
}

// ---------------- K6: output projection + identity (scalar, 67 MFLOP) ----------------
__global__ void k_outproj(const float* __restrict__ oattn, const float* __restrict__ Wout,
                          const float* __restrict__ bout, const float* __restrict__ mq,
                          float* __restrict__ out) {
  int idx = blockIdx.x * blockDim.x + threadIdx.x;  // (q*4 + b)*256 + d
  if (idx >= KQ*BATCH*DIMX) return;
  int d = idx & 255; int b = (idx >> 8) & 3; int q = idx >> 10;
  float acc = bout[d];
  for (int g = 0; g < GG; ++g) {
    const float* src = oattn + (((size_t)b*GG + g)*KQ + q)*DH;
    const float* wrow = Wout + (size_t)(g*DH)*DIMX + d;
    for (int dh = 0; dh < DH; ++dh) acc += src[dh] * wrow[(size_t)dh*DIMX];
  }
  out[idx] = mq[idx] + acc;
}

extern "C" void kernel_launch(void* const* d_in, const int* in_sizes, int n_in,
                              void* d_out, int out_size, void* d_ws, size_t ws_size,
                              hipStream_t stream) {
  (void)in_sizes; (void)n_in; (void)out_size; (void)ws_size;
  const float* mq   = (const float*)d_in[0];
  const float* bev  = (const float*)d_in[1];
  const float* Wtk  = (const float*)d_in[2];
  const float* Wdw  = (const float*)d_in[3];
  const float* bdw  = (const float*)d_in[4];
  const float* Wpw  = (const float*)d_in[5];
  const float* Wp1  = (const float*)d_in[6];
  const float* bp1  = (const float*)d_in[7];
  const float* Wp2  = (const float*)d_in[8];
  const float* bp2  = (const float*)d_in[9];
  const float* Wp3  = (const float*)d_in[10];
  const float* bp3  = (const float*)d_in[11];
  const float* Wp4  = (const float*)d_in[12];
  const float* bp4  = (const float*)d_in[13];
  const float* Wq   = (const float*)d_in[14];
  const float* bq   = (const float*)d_in[15];
  const float* Wk   = (const float*)d_in[16];
  const float* Wv   = (const float*)d_in[17];
  const float* Wout = (const float*)d_in[18];
  const float* bout = (const float*)d_in[19];
  float* out = (float*)d_out;

  size_t cur = 0;
  char* wsb = (char*)d_ws;
  auto alloc = [&](size_t bytes) -> void* {
    size_t a = (cur + 255) & ~(size_t)255;
    cur = a + bytes;
    return (void*)(wsb + a);
  };
  float* vgrid = (float*)alloc((size_t)BGC*NPIX*2*sizeof(float));       // 1 MB
  h16* W2h   = (h16*)alloc((size_t)GG*OUT_G*KTOT*sizeof(h16));          // 1.2 MB
  h16* Wkh   = (h16*)alloc((size_t)GG*OUT_G*IN_G*sizeof(h16));
  h16* Wvh   = (h16*)alloc((size_t)GG*OUT_G*IN_G*sizeof(h16));
  h16* Wp2T  = (h16*)alloc((size_t)POS_H*POS_H*sizeof(h16));
  h16* Wp3T  = (h16*)alloc((size_t)POS_H*POS_H*sizeof(h16));
  h16* Wp4T  = (h16*)alloc((size_t)IN_G*POS_H*sizeof(h16));
  h16* qh    = (h16*)alloc((size_t)BGC*KQ*DH*sizeof(h16));
  h16* kh    = (h16*)alloc((size_t)BGC*NPIX*DH*sizeof(h16));            // 16.8 MB
  h16* vh    = (h16*)alloc((size_t)BGC*NPIX*DH*sizeof(h16));            // 16.8 MB
  float* opart = (float*)alloc((size_t)BGC*NPART*KQ*DH*sizeof(float));  // 4 MB
  float* mlb   = (float*)alloc((size_t)BGC*NPART*KQ*2*sizeof(float));
  float* oattn = (float*)alloc((size_t)BGC*KQ*DH*sizeof(float));

  k_prep<<<dim3(512), dim3(256), 0, stream>>>(Wtk, Wdw, Wk, Wv, Wp2, Wp3, Wp4,
                                              W2h, Wkh, Wvh, Wp2T, Wp3T, Wp4T);
  k_offsets<<<dim3(256, BGC), dim3(128), 0, stream>>>(bev, W2h, bdw, Wpw, vgrid);
  k_sample_kv<<<dim3(HOW, BGC), dim3(128), 0, stream>>>(bev, vgrid, Wp1, bp1, bp2, bp3, bp4,
                                                        Wp2T, Wp3T, Wp4T, Wkh, Wvh, kh, vh);
  k_qproj<<<dim3(512), dim3(256), 0, stream>>>(mq, Wq, bq, qh);
  k_attn<<<dim3(BGC, NPART), dim3(128), 0, stream>>>(qh, kh, vh, opart, mlb);
  k_attn_reduce<<<dim3(512), dim3(256), 0, stream>>>(opart, mlb, oattn);
  k_outproj<<<dim3(256), dim3(256), 0, stream>>>(oattn, Wout, bout, mq, out);
}
